// DocEncoder_79551384256855
// MI455X (gfx1250) — compile-verified, matrix operand order checked by reading, not run on hardware
//
#include <hip/hip_runtime.h>
#include <hip/hip_bf16.h>

// ---------------- types ----------------
typedef __bf16 bf16;
typedef __bf16 v8bf  __attribute__((ext_vector_type(8)));
typedef __bf16 v16bf __attribute__((ext_vector_type(16)));
typedef float  v8f   __attribute__((ext_vector_type(8)));

union Frag { v16bf v; v8bf h[2]; };

__device__ __forceinline__ v8f wmma_bf16(Frag a, Frag b, v8f c) {
    // D(16x16,f32) = A(16x32,bf16) x B(32x16,bf16) + C
    return __builtin_amdgcn_wmma_f32_16x16x32_bf16(
        /*neg_a=*/false, a.v, /*neg_b=*/false, b.v,
        /*c_mod=*/(short)0, c, /*reuse_a=*/false, /*reuse_b=*/false);
}

__device__ __forceinline__ float sigm(float x) { return 1.0f / (1.0f + expf(-x)); }

// ---------------- prep kernels ----------------
// gather embedding rows: fp32 [T,300] and bf16 [T,320] (K padded with zeros)
__global__ void gather_embed_k(const int* __restrict__ tok,
                               const float* __restrict__ table,
                               float* __restrict__ embF, bf16* __restrict__ embH, int T) {
    int idx = blockIdx.x * blockDim.x + threadIdx.x;
    if (idx >= T * 320) return;
    int t = idx / 320, e = idx - t * 320;
    if (e < 300) {
        float v = table[(size_t)tok[t] * 300 + e];
        embF[(size_t)t * 300 + e] = v;
        embH[idx] = (bf16)v;
    } else {
        embH[idx] = (bf16)0.0f;
    }
}

// fp32 [R,C] -> bf16 [Rp,Cp], zero padded
__global__ void convert_pad_k(const float* __restrict__ src, bf16* __restrict__ dst,
                              int R, int C, int Rp, int Cp) {
    int idx = blockIdx.x * blockDim.x + threadIdx.x;
    if (idx >= Rp * Cp) return;
    int r = idx / Cp, c = idx - r * Cp;
    dst[idx] = (r < R && c < C) ? (bf16)src[(size_t)r * C + c] : (bf16)0.0f;
}

__global__ void pad_bias_k(const float* __restrict__ src, float* __restrict__ dst, int R, int Rp) {
    int idx = blockIdx.x * blockDim.x + threadIdx.x;
    if (idx >= Rp) return;
    dst[idx] = (idx < R) ? src[idx] : 0.0f;
}

// W_hh [1024,256] -> W_hhT [256,1024]
__global__ void transpose_whh_k(const float* __restrict__ src, float* __restrict__ dst) {
    int idx = blockIdx.x * blockDim.x + threadIdx.x;
    if (idx >= 256 * 1024) return;
    int k = idx >> 10, j = idx & 1023;
    dst[(size_t)k * 1024 + j] = src[(size_t)j * 256 + k];
}

// ---------------- WMMA GEMM ----------------
// out[M,N] = A[M,K] @ B[N,K]^T (+bias) (+relu). A,B bf16, K = KSTEPS*32.
// B padded to full block column coverage so the inner loop is branch-free.
// Rolled 2-stage ping-pong k-loop (slot indices are compile-time constants in
// the body -> zero register copies, bounded live set -> no spills).
// block = 128 threads (4 waves), each wave computes 16(M) x 64(N).
// grid = (M/16, ceil(Ncover/256))
template <int KSTEPS>
__global__ void gemm_wmma_k(const bf16* __restrict__ A, int lda,
                            const bf16* __restrict__ B, int ldb,
                            const float* __restrict__ bias,
                            float* __restrict__ outF, bf16* __restrict__ outH, int ldo,
                            int N, int Npad, int doRelu) {
    const int lane = threadIdx.x & 31;
    const int wave = threadIdx.x >> 5;
    const int m0   = blockIdx.x * 16;
    const int nbase = blockIdx.y * 256 + wave * 64;
    const int koff = (lane >> 4) * 8;   // K-half selector per ISA 16-bit layout

    const bf16* ap  = A + (size_t)(m0 + (lane & 15)) * lda + koff;
    const bf16* bp0 = B + (size_t)(nbase + (lane & 15)) * ldb + koff;

    v8f acc[4];
#pragma unroll
    for (int t = 0; t < 4; ++t)
#pragma unroll
        for (int r = 0; r < 8; ++r) acc[t][r] = 0.0f;

    Frag a[2];
    Frag b[2][4];
    auto loadStep = [&](int s, int slot) {
        const int kk = s * 32;
        a[slot].h[0] = *reinterpret_cast<const v8bf*>(ap + kk);
        a[slot].h[1] = *reinterpret_cast<const v8bf*>(ap + kk + 16);
#pragma unroll
        for (int t = 0; t < 4; ++t) {
            const bf16* bp = bp0 + (size_t)(t * 16) * ldb + kk;
            b[slot][t].h[0] = *reinterpret_cast<const v8bf*>(bp);
            b[slot][t].h[1] = *reinterpret_cast<const v8bf*>(bp + 16);
        }
    };

    loadStep(0, 0);
#pragma unroll 1
    for (int s = 0; s < KSTEPS - 1; s += 2) {
        loadStep(s + 1, 1);
#pragma unroll
        for (int t = 0; t < 4; ++t) acc[t] = wmma_bf16(a[0], b[0][t], acc[t]);
        if (s + 2 < KSTEPS) loadStep(s + 2, 0);
#pragma unroll
        for (int t = 0; t < 4; ++t) acc[t] = wmma_bf16(a[1], b[1][t], acc[t]);
    }
    if (KSTEPS & 1) {  // tail step already resident in slot 0
#pragma unroll
        for (int t = 0; t < 4; ++t) acc[t] = wmma_bf16(a[0], b[0][t], acc[t]);
    }

    const int ncol = lane & 15;
    const int mhi  = (lane < 16) ? 0 : 8;
#pragma unroll
    for (int t = 0; t < 4; ++t) {
        int gn = nbase + t * 16 + ncol;
        if (gn >= Npad) continue;
        bool valid = gn < N;
        float bv = (bias && valid) ? bias[gn] : 0.0f;
#pragma unroll
        for (int r = 0; r < 8; ++r) {
            int gm = m0 + mhi + r;
            float v = acc[t][r] + bv;
            if (doRelu) v = v > 0.0f ? v : 0.0f;
            if (!valid) v = 0.0f;
            if (outF) outF[(size_t)gm * ldo + gn] = v;
            if (outH) outH[(size_t)gm * ldo + gn] = (bf16)v;
        }
    }
}

// ---------------- BiLSTM recurrence ----------------
// grid = 2 (dir), block = 1024. h kept in LDS; W_hhT streamed from L2.
__global__ void lstm_k(const float* __restrict__ xp_f, const float* __restrict__ xp_b,
                       const float* __restrict__ WtF,  const float* __restrict__ WtB,
                       bf16* __restrict__ hB, int T) {
    const int dir = blockIdx.x;
    const float* xp = dir ? xp_b : xp_f;
    const float* Wt = dir ? WtB  : WtF;
    __shared__ float hs[256];
    __shared__ float zs[1024];
    const int j = threadIdx.x;
    float c = 0.0f;
    if (j < 256) hs[j] = 0.0f;
    __syncthreads();
    for (int step = 0; step < T; ++step) {
        int t = dir ? (T - 1 - step) : step;
        float acc = xp[(size_t)t * 1024 + j];
#pragma unroll 8
        for (int k = 0; k < 256; ++k) acc += Wt[k * 1024 + j] * hs[k];
        zs[j] = acc;
        __syncthreads();
        if (j < 256) {
            float zi = zs[j], zf = zs[j + 256], zg = zs[j + 512], zo = zs[j + 768];
            c = sigm(zf) * c + sigm(zi) * tanhf(zg);
            float hv = sigm(zo) * tanhf(c);
            hs[j] = hv;
            hB[(size_t)t * 512 + dir * 256 + j] = (bf16)hv;
        }
        __syncthreads();
    }
}

// ---------------- attention logit (last 150->1 layer) ----------------
__global__ void logits_k(const bf16* __restrict__ a2, const float* __restrict__ w3,
                         const float* __restrict__ b3, float* __restrict__ logits, int T) {
    int t = blockIdx.x * blockDim.x + threadIdx.x;
    if (t >= T) return;
    float acc = b3[0];
    for (int k = 0; k < 150; ++k) acc += (float)a2[(size_t)t * 160 + k] * w3[k];
    logits[t] = acc;
}

// ---------------- fused span scorer ----------------
// block = 320 threads (10 waves) handles 16 spans of one window size n.
// g tile [16 x 1344] bf16 built in LDS; 3-layer MLP via WMMA, ONE n-tile per
// wave (no outer tile loop -> nothing for LICM to hoist/spill).
__global__ void span_k(int n, int Wn, int base,
                       const float* __restrict__ logits, const float* __restrict__ embF,
                       const bf16* __restrict__ hB,
                       const bf16* __restrict__ sw1b, const float* __restrict__ sb1p,
                       const bf16* __restrict__ sw2b, const float* __restrict__ sb2p,
                       const float* __restrict__ sw3,  const float* __restrict__ sb3,
                       float* __restrict__ out) {
    __shared__ __align__(16) bf16 g[16][1344];
    __shared__ __align__(16) bf16 o1[16][160];
    __shared__ __align__(16) bf16 o2[16][160];
    __shared__ float aw[16][10];

    const int tid = threadIdx.x;
    const int s0 = blockIdx.x * 16;

    // h[start] (cols 0..511), h[end] (cols 512..1023)
    for (int idx = tid; idx < 16 * 512; idx += 320) {
        int s = idx >> 9, col = idx & 511;
        int w = s0 + s;
        if (w < Wn) {
            g[s][col]       = hB[(size_t)w * 512 + col];
            g[s][512 + col] = hB[(size_t)(w + n - 1) * 512 + col];
        } else {
            g[s][col] = (bf16)0.0f;
            g[s][512 + col] = (bf16)0.0f;
        }
    }
    // softmax weights over window, plus tail columns (size feature + zero pad)
    if (tid < 16) {
        int s = tid, w = s0 + s;
        if (w < Wn) {
            float m = -1e30f;
            for (int j = 0; j < n; ++j) m = fmaxf(m, logits[w + j]);
            float e[10], sum = 0.0f;
            for (int j = 0; j < n; ++j) { e[j] = expf(logits[w + j] - m); sum += e[j]; }
            for (int j = 0; j < n; ++j) aw[s][j] = e[j] / sum;
        } else {
            for (int j = 0; j < n; ++j) aw[s][j] = 0.0f;
        }
        g[s][1324] = (bf16)(float)n;
        for (int c2 = 1325; c2 < 1344; ++c2) g[s][c2] = (bf16)0.0f;
    }
    __syncthreads();
    // attention-pooled embedding (cols 1024..1323)
    if (tid < 256) {
        int s = tid >> 4;
        int w = s0 + s;
        for (int e = (tid & 15); e < 300; e += 16) {
            float acc = 0.0f;
            if (w < Wn)
                for (int j = 0; j < n; ++j) acc += aw[s][j] * embF[(size_t)(w + j) * 300 + e];
            g[s][1024 + e] = (bf16)acc;
        }
    }
    __syncthreads();

    const int lane = tid & 31, wave = tid >> 5;   // wave = n-tile id, 0..9
    const int koff = (lane >> 4) * 8;
    const int mr = lane & 15;
    const int mhi = (lane < 16) ? 0 : 8;
    const int gn = wave * 16 + mr;

    // layer1: [16,1344] @ sw1^T -> relu -> o1 [16,160]  (42 k-steps, ping-pong)
    {
        v8f acc;
#pragma unroll
        for (int r = 0; r < 8; ++r) acc[r] = 0.0f;
        const bf16* bp0 = sw1b + (size_t)gn * 1344 + koff;
        Frag a[2], b[2];
        auto load1 = [&](int s, int slot) {
            const int kk = s * 32;
            const bf16* apl = &g[mr][kk + koff];
            a[slot].h[0] = *reinterpret_cast<const v8bf*>(apl);
            a[slot].h[1] = *reinterpret_cast<const v8bf*>(apl + 16);
            b[slot].h[0] = *reinterpret_cast<const v8bf*>(bp0 + kk);
            b[slot].h[1] = *reinterpret_cast<const v8bf*>(bp0 + kk + 16);
        };
        load1(0, 0);
#pragma unroll 1
        for (int s = 0; s < 41; s += 2) {   // 42 steps, even
            load1(s + 1, 1);
            acc = wmma_bf16(a[0], b[0], acc);
            if (s + 2 < 42) load1(s + 2, 0);
            acc = wmma_bf16(a[1], b[1], acc);
        }
        float bv = sb1p[gn];
#pragma unroll
        for (int r = 0; r < 8; ++r) {
            float v = acc[r] + bv;
            v = v > 0.0f ? v : 0.0f;
            o1[mhi + r][gn] = (bf16)v;
        }
    }
    __syncthreads();

    // layer2: [16,160] @ sw2^T -> relu -> o2 [16,160]  (5 k-steps, ping-pong + tail)
    {
        v8f acc;
#pragma unroll
        for (int r = 0; r < 8; ++r) acc[r] = 0.0f;
        const bf16* bp0 = sw2b + (size_t)gn * 160 + koff;
        Frag a[2], b[2];
        auto load2 = [&](int s, int slot) {
            const int kk = s * 32;
            const bf16* apl = &o1[mr][kk + koff];
            a[slot].h[0] = *reinterpret_cast<const v8bf*>(apl);
            a[slot].h[1] = *reinterpret_cast<const v8bf*>(apl + 16);
            b[slot].h[0] = *reinterpret_cast<const v8bf*>(bp0 + kk);
            b[slot].h[1] = *reinterpret_cast<const v8bf*>(bp0 + kk + 16);
        };
        load2(0, 0);
#pragma unroll 1
        for (int s = 0; s < 4; s += 2) {    // 5 steps: pairs then tail
            load2(s + 1, 1);
            acc = wmma_bf16(a[0], b[0], acc);
            if (s + 2 < 5) load2(s + 2, 0);
            acc = wmma_bf16(a[1], b[1], acc);
        }
        acc = wmma_bf16(a[0], b[0], acc);   // tail (step 4 resident in slot 0)
        float bv = sb2p[gn];
#pragma unroll
        for (int r = 0; r < 8; ++r) {
            float v = acc[r] + bv;
            v = v > 0.0f ? v : 0.0f;
            o2[mhi + r][gn] = (bf16)v;
        }
    }
    __syncthreads();

    // layer3: 150 -> 1
    if (tid < 16) {
        int s = tid, w = s0 + s;
        if (w < Wn) {
            float acc = sb3[0];
            for (int k = 0; k < 150; ++k) acc += (float)o2[s][k] * sw3[k];
            out[base + w] = acc;
        }
    }
}

// ---------------- host launcher ----------------
static inline int cdiv(int a, int b) { return (a + b - 1) / b; }

extern "C" void kernel_launch(void* const* d_in, const int* in_sizes, int n_in,
                              void* d_out, int out_size, void* d_ws, size_t ws_size,
                              hipStream_t stream) {
    (void)n_in; (void)out_size; (void)ws_size;
    const int T = in_sizes[0];           // 4096
    const int E = 300, Ep = 320, FOURH = 1024, S = 512;
    const int Acol = 150, Ap = 160, Apad = 256, G = 1325, Gp = 1344;

    const int*   tok   = (const int*)  d_in[0];
    const float* table = (const float*)d_in[1];
    const float* Wih_f = (const float*)d_in[2];
    const float* Whh_f = (const float*)d_in[3];
    const float* b_f   = (const float*)d_in[4];
    const float* Wih_b = (const float*)d_in[5];
    const float* Whh_b = (const float*)d_in[6];
    const float* b_b   = (const float*)d_in[7];
    const float* aw1   = (const float*)d_in[8];
    const float* ab1   = (const float*)d_in[9];
    const float* aw2   = (const float*)d_in[10];
    const float* ab2   = (const float*)d_in[11];
    const float* aw3   = (const float*)d_in[12];
    const float* ab3   = (const float*)d_in[13];
    const float* sw1   = (const float*)d_in[14];
    const float* sb1   = (const float*)d_in[15];
    const float* sw2   = (const float*)d_in[16];
    const float* sb2   = (const float*)d_in[17];
    const float* sw3   = (const float*)d_in[18];
    const float* sb3   = (const float*)d_in[19];
    float* out = (float*)d_out;

    // workspace carve
    char* p = (char*)d_ws;
    auto alloc = [&](size_t bytes) -> void* {
        void* r = (void*)p;
        p += (bytes + 255) & ~(size_t)255;
        return r;
    };
    float* embF  = (float*)alloc((size_t)T * E * 4);
    bf16*  embH  = (bf16*) alloc((size_t)T * Ep * 2);
    bf16*  wihfH = (bf16*) alloc((size_t)FOURH * Ep * 2);
    bf16*  wihbH = (bf16*) alloc((size_t)FOURH * Ep * 2);
    float* whhtF = (float*)alloc((size_t)256 * 1024 * 4);
    float* whhtB = (float*)alloc((size_t)256 * 1024 * 4);
    float* xpF   = (float*)alloc((size_t)T * FOURH * 4);
    float* xpB   = (float*)alloc((size_t)T * FOURH * 4);
    bf16*  hB    = (bf16*) alloc((size_t)T * S * 2);
    bf16*  aw1H  = (bf16*) alloc((size_t)Apad * S * 2);   // rows padded to full 256-col coverage
    bf16*  aw2H  = (bf16*) alloc((size_t)Apad * Ap * 2);  // rows padded to full 256-col coverage
    bf16*  sw1H  = (bf16*) alloc((size_t)Ap * Gp * 2);
    bf16*  sw2H  = (bf16*) alloc((size_t)Ap * Ap * 2);
    float* ab1p  = (float*)alloc((size_t)Ap * 4);
    float* ab2p  = (float*)alloc((size_t)Ap * 4);
    float* sb1p  = (float*)alloc((size_t)Ap * 4);
    float* sb2p  = (float*)alloc((size_t)Ap * 4);
    bf16*  a1    = (bf16*) alloc((size_t)T * Ap * 2);
    bf16*  a2    = (bf16*) alloc((size_t)T * Ap * 2);
    float* logit = (float*)alloc((size_t)T * 4);

    // ---- prep ----
    gather_embed_k<<<cdiv(T * Ep, 256), 256, 0, stream>>>(tok, table, embF, embH, T);
    convert_pad_k<<<cdiv(FOURH * Ep, 256), 256, 0, stream>>>(Wih_f, wihfH, FOURH, E, FOURH, Ep);
    convert_pad_k<<<cdiv(FOURH * Ep, 256), 256, 0, stream>>>(Wih_b, wihbH, FOURH, E, FOURH, Ep);
    convert_pad_k<<<cdiv(Apad * S, 256), 256, 0, stream>>>(aw1, aw1H, Acol, S, Apad, S);
    convert_pad_k<<<cdiv(Apad * Ap, 256), 256, 0, stream>>>(aw2, aw2H, Acol, Acol, Apad, Ap);
    convert_pad_k<<<cdiv(Ap * Gp, 256), 256, 0, stream>>>(sw1, sw1H, Acol, G, Ap, Gp);
    convert_pad_k<<<cdiv(Ap * Ap, 256), 256, 0, stream>>>(sw2, sw2H, Acol, Acol, Ap, Ap);
    pad_bias_k<<<1, 256, 0, stream>>>(ab1, ab1p, Acol, Ap);
    pad_bias_k<<<1, 256, 0, stream>>>(ab2, ab2p, Acol, Ap);
    pad_bias_k<<<1, 256, 0, stream>>>(sb1, sb1p, Acol, Ap);
    pad_bias_k<<<1, 256, 0, stream>>>(sb2, sb2p, Acol, Ap);
    transpose_whh_k<<<cdiv(256 * 1024, 256), 256, 0, stream>>>(Whh_f, whhtF);
    transpose_whh_k<<<cdiv(256 * 1024, 256), 256, 0, stream>>>(Whh_b, whhtB);

    // ---- input projections: xp = emb @ W_ih^T + b  (M=T, N=1024, K=320) ----
    {
        dim3 grid(T / 16, cdiv(FOURH, 256));
        gemm_wmma_k<10><<<grid, 128, 0, stream>>>(embH, Ep, wihfH, Ep, b_f,
                                                  xpF, nullptr, FOURH, FOURH, FOURH, 0);
        gemm_wmma_k<10><<<grid, 128, 0, stream>>>(embH, Ep, wihbH, Ep, b_b,
                                                  xpB, nullptr, FOURH, FOURH, FOURH, 0);
    }

    // ---- BiLSTM recurrence (sequential critical path) ----
    lstm_k<<<2, 1024, 0, stream>>>(xpF, xpB, whhtF, whhtB, hB, T);

    // ---- attention MLP: h[T,512] -> 150 -> 150 -> 1 ----
    {
        dim3 grid(T / 16, 1);
        gemm_wmma_k<16><<<grid, 128, 0, stream>>>(hB, S, aw1H, S, ab1p,
                                                  nullptr, a1, Ap, Acol, Ap, 1);
        gemm_wmma_k<5><<<grid, 128, 0, stream>>>(a1, Ap, aw2H, Ap, ab2p,
                                                 nullptr, a2, Ap, Acol, Ap, 1);
    }
    logits_k<<<cdiv(T, 256), 256, 0, stream>>>(a2, aw3, ab3, logit, T);

    // ---- fused span scorer, window sizes 1..10 ----
    int base = 0;
    for (int n = 1; n <= 10; ++n) {
        int Wn = T - n + 1;
        span_k<<<cdiv(Wn, 16), 320, 0, stream>>>(n, Wn, base, logit, embF, hB,
                                                 sw1H, sb1p, sw2H, sb2p, sw3, sb3, out);
        base += Wn;
    }
}
